// GraphNetFeaturesPointNet_83614423318873
// MI455X (gfx1250) — matrix operand order repaired and checked by reading, not compiled, not run on hardware
//
#include <hip/hip_runtime.h>

// ---------------- problem constants (match reference setup_inputs) ----------
#define N_NODES 50000
#define N_EDGES 1600000LL
#define NBATCH  64
#define F_IN    1024
#define H1      1128
#define H2      1256
#define FC1     512
#define FC2     256
#define EMB     64
#define EPS_BN  1e-5f

// ---------------- CDNA5 async global->LDS support (guarded) ----------------
#if defined(__has_builtin)
#  if __has_builtin(__builtin_amdgcn_global_load_async_to_lds_b128) && \
      __has_builtin(__builtin_amdgcn_s_wait_asynccnt)
#    define USE_ASYNC 1
#  endif
#endif
#ifndef USE_ASYNC
#  define USE_ASYNC 0
#endif

typedef int v4i __attribute__((ext_vector_type(4)));
#if USE_ASYNC
typedef __attribute__((address_space(1))) v4i g_v4i;  // global int4
typedef __attribute__((address_space(3))) v4i l_v4i;  // LDS int4
#endif

// ---------------- types ----------------
typedef __attribute__((ext_vector_type(16))) __bf16 bf16x16;
typedef __attribute__((ext_vector_type(8)))  float  floatx8;

struct __align__(16) U4 { unsigned x, y, z, w; };
union Frag { bf16x16 v; U4 q[2]; };

__device__ __forceinline__ unsigned short f2bf(float f) {
  // round-to-nearest-even f32 -> bf16
  unsigned u = __float_as_uint(f);
  u += 0x7fffu + ((u >> 16) & 1u);
  return (unsigned short)(u >> 16);
}

// ---------------- utility kernels ----------------
__global__ void k_fill_f32(float* __restrict__ p, long long n, float v) {
  long long t = (long long)blockIdx.x * blockDim.x + threadIdx.x;
  if (t < n) p[t] = v;
}

__global__ void k_deg_accum(const long long* __restrict__ ei, float* __restrict__ deg, long long E) {
  long long t = (long long)blockIdx.x * blockDim.x + threadIdx.x;
  if (t < E) {
    int dst = (int)ei[E + t];
    atomicAdd(&deg[dst], 1.0f);
  }
}

__global__ void k_f32_to_bf16(const float* __restrict__ in, unsigned short* __restrict__ out, long long n) {
  long long t = (long long)blockIdx.x * blockDim.x + threadIdx.x;
  if (t < n) out[t] = f2bf(in[t]);
}

// W [K][Nout] f32  ->  Wt [Nout][K] bf16 (K-major so GEMM B-tiles stage contiguously)
__global__ void k_w_to_bf16_t(const float* __restrict__ Wsrc, unsigned short* __restrict__ Wt,
                              int K, int Nout) {
  long long t = (long long)blockIdx.x * blockDim.x + threadIdx.x;
  if (t >= (long long)K * Nout) return;
  int n = (int)(t / K);
  int k = (int)(t % K);
  Wt[t] = f2bf(Wsrc[(long long)k * Nout + n]);
}

// ---------------- WMMA bf16 GEMM: C[M,Nout] = A[M,K] * Bt[Nout,K]^T ---------
// Block tile 128x128, K-step 32, double-buffered LDS (async global->LDS when
// available). 8 waves (wave32): 4 M-groups x 2 N-groups, each wave 32x64 =
// 2x4 tiles of 16x16 via v_wmma_f32_16x16x32_bf16.
#define LDT 40   // padded LDS row stride (shorts): 80B rows keep 16B alignment

// Stage a 128-row x 32-K bf16 tile (row-major, row stride K) into LDS.
__device__ __forceinline__ void stage_tile_128x32(
    const unsigned short* __restrict__ src, unsigned short* dst,
    int row0, int k0, int R, int K, int tid) {
  for (int c = tid; c < 512; c += 256) {
    const int row = c >> 2;          // 0..127
    const int kc  = (c & 3) << 3;    // 0,8,16,24
    const int gr  = row0 + row;
    const int gk  = k0 + kc;
    unsigned short* lp = dst + row * LDT + kc;
    if (gr < R && gk + 8 <= K) {
#if USE_ASYNC
      __builtin_amdgcn_global_load_async_to_lds_b128(
          (g_v4i*)(unsigned long long)(src + (long long)gr * K + gk),
          (l_v4i*)(unsigned long long)lp, 0, 0);
#else
      *(U4*)lp = *(const U4*)(src + (long long)gr * K + gk);
#endif
    } else {
      __align__(16) unsigned short tmp[8] = {0, 0, 0, 0, 0, 0, 0, 0};
      if (gr < R) {
        for (int j = 0; j < 8; ++j)
          if (gk + j < K) tmp[j] = src[(long long)gr * K + gk + j];
      }
      *(U4*)lp = *(const U4*)tmp;
    }
  }
}

__global__ __launch_bounds__(256)
void k_gemm_bf16(const unsigned short* __restrict__ A,   // [M][K]
                 const unsigned short* __restrict__ Bt,  // [Nout][K]
                 float* __restrict__ C,
                 int M, int K, int Nout) {
  __shared__ unsigned short lA[2][128 * LDT];  // [m][k]
  __shared__ unsigned short lB[2][128 * LDT];  // [n][k]

  const int tid   = threadIdx.x;
  const int n0    = blockIdx.x * 128;
  const int m0    = blockIdx.y * 128;
  const int lane  = tid & 31;
  const int wave  = tid >> 5;
  const int hl    = lane >> 4;   // half-wave selects contiguous K-half (k = 16*hl + s)
  const int lr    = lane & 15;
  const int waveM = wave & 3;
  const int waveN = wave >> 2;

  floatx8 acc[2][4];
  const floatx8 zf = {0.f, 0.f, 0.f, 0.f, 0.f, 0.f, 0.f, 0.f};
  for (int mt = 0; mt < 2; ++mt)
    for (int nt = 0; nt < 4; ++nt) acc[mt][nt] = zf;

  // prologue: stage first K-tile into buffer 0
  stage_tile_128x32(A,  lA[0], m0, 0, M,    K, tid);
  stage_tile_128x32(Bt, lB[0], n0, 0, Nout, K, tid);
#if USE_ASYNC
  __builtin_amdgcn_s_wait_asynccnt(0);
#endif
  __syncthreads();

  int buf = 0;
  for (int k0 = 0; k0 < K; k0 += 32) {
    // stage next tile into the other buffer while computing on this one
    if (k0 + 32 < K) {
      stage_tile_128x32(A,  lA[buf ^ 1], m0, k0 + 32, M,    K, tid);
      stage_tile_128x32(Bt, lB[buf ^ 1], n0, k0 + 32, Nout, K, tid);
    }

    const unsigned short* sA = lA[buf];
    const unsigned short* sB = lB[buf];

    Frag fa[2];
#pragma unroll
    for (int mt = 0; mt < 2; ++mt) {
      int row = waveM * 32 + mt * 16 + lr;
      // A 16x32 layout: contiguous K run [hl*16 .. hl*16+15]
      fa[mt].q[0] = *(const U4*)&sA[row * LDT + hl * 16];
      fa[mt].q[1] = *(const U4*)&sA[row * LDT + hl * 16 + 8];
    }
    Frag fb[4];
#pragma unroll
    for (int nt = 0; nt < 4; ++nt) {
      int col = waveN * 64 + nt * 16 + lr;
      // B 32x16 layout: lane = N col; contiguous K run [hl*16 .. hl*16+15]
      fb[nt].q[0] = *(const U4*)&sB[col * LDT + hl * 16];
      fb[nt].q[1] = *(const U4*)&sB[col * LDT + hl * 16 + 8];
    }
#pragma unroll
    for (int mt = 0; mt < 2; ++mt)
#pragma unroll
      for (int nt = 0; nt < 4; ++nt)
        acc[mt][nt] = __builtin_amdgcn_wmma_f32_16x16x32_bf16(
            false, fa[mt].v, false, fb[nt].v, (short)0, acc[mt][nt], false, false);

#if USE_ASYNC
    __builtin_amdgcn_s_wait_asynccnt(0);
#endif
    __syncthreads();
    buf ^= 1;
  }

  // epilogue: D layout: VGPR r -> row (r + 8*half), lane%16 -> col
  for (int mt = 0; mt < 2; ++mt) {
    for (int nt = 0; nt < 4; ++nt) {
      int gcol = n0 + waveN * 64 + nt * 16 + lr;
      if (gcol >= Nout) continue;
      for (int r = 0; r < 8; ++r) {
        int grow = m0 + waveM * 32 + mt * 16 + hl * 8 + r;
        if (grow < M) C[(long long)grow * Nout + gcol] = acc[mt][nt][r];
      }
    }
  }
}

// ---------------- graph propagation ----------------
__global__ void k_propagate(const float* __restrict__ h, const long long* __restrict__ ei,
                            const float* __restrict__ deg, float* __restrict__ agg,
                            int H, long long E) {
  long long e = blockIdx.x;
  int src = (int)ei[e];
  int dst = (int)ei[E + e];
  float coef = rsqrtf(deg[src]) * rsqrtf(deg[dst]);
  const float* hs = h + (long long)src * H;
  float* ad = agg + (long long)dst * H;
  for (int f = threadIdx.x; f < H; f += blockDim.x)
    atomicAdd(&ad[f], hs[f] * coef);
}

__global__ void k_selfloop_bias(float* __restrict__ agg, const float* __restrict__ h,
                                const float* __restrict__ deg, const float* __restrict__ bias,
                                int n, int H) {
  long long t = (long long)blockIdx.x * blockDim.x + threadIdx.x;
  if (t >= (long long)n * H) return;
  int i = (int)(t / H);
  int f = (int)(t % H);
  agg[t] += h[t] / deg[i] + bias[f];
}

// ---------------- batchnorm (training stats, biased var) ----------------
__global__ void k_colstats(const float* __restrict__ a, float* __restrict__ s,
                           float* __restrict__ ss, int rows, int H, int rowsPerBlock) {
  int col = blockIdx.x * blockDim.x + threadIdx.x;
  if (col >= H) return;
  int r0 = blockIdx.y * rowsPerBlock;
  int r1 = min(rows, r0 + rowsPerBlock);
  float sum = 0.f, sq = 0.f;
  for (int r = r0; r < r1; ++r) {
    float v = a[(long long)r * H + col];
    sum += v; sq += v * v;
  }
  atomicAdd(&s[col], sum);
  atomicAdd(&ss[col], sq);
}

__global__ void k_bn_relu_bf16(const float* __restrict__ a, const float* __restrict__ s,
                               const float* __restrict__ ss, const float* __restrict__ g,
                               const float* __restrict__ be, unsigned short* __restrict__ out,
                               int rows, int H) {
  long long t = (long long)blockIdx.x * blockDim.x + threadIdx.x;
  if (t >= (long long)rows * H) return;
  int f = (int)(t % H);
  float m   = s[f] / rows;
  float var = ss[f] / rows - m * m;
  float v = g[f] * (a[t] - m) * rsqrtf(var + EPS_BN) + be[f];
  out[t] = f2bf(fmaxf(v, 0.f));
}

__global__ void k_bn_relu_f32(const float* __restrict__ a, const float* __restrict__ s,
                              const float* __restrict__ ss, const float* __restrict__ g,
                              const float* __restrict__ be, float* __restrict__ out,
                              int rows, int H) {
  long long t = (long long)blockIdx.x * blockDim.x + threadIdx.x;
  if (t >= (long long)rows * H) return;
  int f = (int)(t % H);
  float m   = s[f] / rows;
  float var = ss[f] / rows - m * m;
  float v = g[f] * (a[t] - m) * rsqrtf(var + EPS_BN) + be[f];
  out[t] = fmaxf(v, 0.f);
}

// ---------------- global max pool (values >= 0 => uint-bit atomicMax) -------
__global__ void k_pool_max(const float* __restrict__ h, const long long* __restrict__ batch,
                           float* __restrict__ pooled, int n, int H) {
  long long t = (long long)blockIdx.x * blockDim.x + threadIdx.x;
  if (t >= (long long)n * H) return;
  int i = (int)(t / H);
  int f = (int)(t % H);
  int b = (int)batch[i];
  atomicMax((unsigned int*)&pooled[(long long)b * H + f], __float_as_uint(h[t]));
}

// ---------------- small MLP head ----------------
__global__ void k_fc(const float* __restrict__ in, const float* __restrict__ Wg,
                     const float* __restrict__ bias, float* __restrict__ out,
                     int rows, int K, int Nout) {
  int t = blockIdx.x * blockDim.x + threadIdx.x;
  if (t >= rows * Nout) return;
  int b = t / Nout, o = t % Nout;
  float acc = bias[o];
  const float* ib = in + (long long)b * K;
  for (int k = 0; k < K; ++k) acc += ib[k] * Wg[(long long)k * Nout + o];
  out[t] = acc;
}

__global__ void k_bn_relu_small(float* __restrict__ z, const float* __restrict__ g,
                                const float* __restrict__ be, int rows, int H) {
  int col = blockIdx.x, t = threadIdx.x;  // blockDim == rows == 64
  float v = z[(long long)t * H + col];
  __shared__ float s1[64], s2[64];
  s1[t] = v; s2[t] = v * v;
  __syncthreads();
  for (int o = 32; o > 0; o >>= 1) {
    if (t < o) { s1[t] += s1[t + o]; s2[t] += s2[t + o]; }
    __syncthreads();
  }
  float m   = s1[0] / rows;
  float var = s2[0] / rows - m * m;
  z[(long long)t * H + col] = fmaxf(g[col] * (v - m) * rsqrtf(var + EPS_BN) + be[col], 0.f);
}

__global__ void k_l2norm(const float* __restrict__ z, float* __restrict__ out) {
  int b = blockIdx.x, t = threadIdx.x;  // blockDim == EMB == 64
  float v = z[b * EMB + t];
  __shared__ float s[EMB];
  s[t] = v * v;
  __syncthreads();
  for (int o = 32; o > 0; o >>= 1) {
    if (t < o) s[t] += s[t + o];
    __syncthreads();
  }
  float norm = fmaxf(sqrtf(s[0]), 1e-12f);
  out[b * EMB + t] = v / norm;
}

// ---------------- launch ----------------
extern "C" void kernel_launch(void* const* d_in, const int* in_sizes, int n_in,
                              void* d_out, int out_size, void* d_ws, size_t ws_size,
                              hipStream_t stream) {
  (void)in_sizes; (void)n_in; (void)out_size; (void)ws_size;

  const float*     x     = (const float*)d_in[0];
  const long long* ei    = (const long long*)d_in[1];
  const long long* batch = (const long long*)d_in[2];
  const float* W1 = (const float*)d_in[3];
  const float* b1 = (const float*)d_in[4];
  const float* g1 = (const float*)d_in[5];
  const float* be1 = (const float*)d_in[6];
  const float* W2 = (const float*)d_in[7];
  const float* b2 = (const float*)d_in[8];
  const float* g2 = (const float*)d_in[9];
  const float* be2 = (const float*)d_in[10];
  const float* Wf1 = (const float*)d_in[11];
  const float* bf1 = (const float*)d_in[12];
  const float* g5 = (const float*)d_in[13];
  const float* be5 = (const float*)d_in[14];
  const float* Wf2 = (const float*)d_in[15];
  const float* bf2 = (const float*)d_in[16];
  const float* g6 = (const float*)d_in[17];
  const float* be6 = (const float*)d_in[18];
  const float* Wf3 = (const float*)d_in[19];
  const float* bf3 = (const float*)d_in[20];
  float* out = (float*)d_out;

  // workspace layout (256B aligned slabs, reused across layers)
  char* w = (char*)d_ws;
  size_t off = 0;
  auto take = [&](size_t bytes) -> char* {
    char* p = w + off;
    off += (bytes + 255) & ~(size_t)255;
    return p;
  };
  float*          deg    = (float*)take((size_t)N_NODES * 4);
  unsigned short* Abf    = (unsigned short*)take((size_t)N_NODES * H1 * 2);  // x_bf16 then h1_bf16
  unsigned short* Wbf    = (unsigned short*)take((size_t)H1 * H2 * 2);       // W1t/W2t bf16 (K-major)
  float*          hbuf   = (float*)take((size_t)N_NODES * H2 * 4);           // GEMM out / BN out
  float*          agg    = (float*)take((size_t)N_NODES * H2 * 4);           // propagation accum
  float*          colsum = (float*)take((size_t)H2 * 4);
  float*          colsq  = (float*)take((size_t)H2 * 4);
  float*          pooled = (float*)take((size_t)NBATCH * H2 * 4);
  float*          z1     = (float*)take((size_t)NBATCH * FC1 * 4);
  float*          z2     = (float*)take((size_t)NBATCH * FC2 * 4);
  float*          z3     = (float*)take((size_t)NBATCH * EMB * 4);

  const int T = 256;
  auto nb = [](long long n, int t) { return (unsigned)((n + t - 1) / t); };

  // degrees (self-loop +1)
  k_fill_f32<<<nb(N_NODES, T), T, 0, stream>>>(deg, N_NODES, 1.0f);
  k_deg_accum<<<nb(N_EDGES, T), T, 0, stream>>>(ei, deg, N_EDGES);

  // ----- layer 1 -----
  k_f32_to_bf16<<<nb((long long)N_NODES * F_IN, T), T, 0, stream>>>(x, Abf, (long long)N_NODES * F_IN);
  k_w_to_bf16_t<<<nb((long long)F_IN * H1, T), T, 0, stream>>>(W1, Wbf, F_IN, H1);
  {
    dim3 g((H1 + 127) / 128, (N_NODES + 127) / 128);
    k_gemm_bf16<<<g, 256, 0, stream>>>(Abf, Wbf, hbuf, N_NODES, F_IN, H1);
  }
  k_fill_f32<<<nb((long long)N_NODES * H1, T), T, 0, stream>>>(agg, (long long)N_NODES * H1, 0.f);
  k_propagate<<<(unsigned)N_EDGES, 256, 0, stream>>>(hbuf, ei, deg, agg, H1, N_EDGES);
  k_selfloop_bias<<<nb((long long)N_NODES * H1, T), T, 0, stream>>>(agg, hbuf, deg, b1, N_NODES, H1);
  k_fill_f32<<<nb(H1, T), T, 0, stream>>>(colsum, H1, 0.f);
  k_fill_f32<<<nb(H1, T), T, 0, stream>>>(colsq, H1, 0.f);
  {
    dim3 g((H1 + 255) / 256, 64);
    k_colstats<<<g, 256, 0, stream>>>(agg, colsum, colsq, N_NODES, H1, (N_NODES + 63) / 64);
  }
  k_bn_relu_bf16<<<nb((long long)N_NODES * H1, T), T, 0, stream>>>(agg, colsum, colsq, g1, be1,
                                                                   Abf, N_NODES, H1);

  // ----- layer 2 -----
  k_w_to_bf16_t<<<nb((long long)H1 * H2, T), T, 0, stream>>>(W2, Wbf, H1, H2);
  {
    dim3 g((H2 + 127) / 128, (N_NODES + 127) / 128);
    k_gemm_bf16<<<g, 256, 0, stream>>>(Abf, Wbf, hbuf, N_NODES, H1, H2);
  }
  k_fill_f32<<<nb((long long)N_NODES * H2, T), T, 0, stream>>>(agg, (long long)N_NODES * H2, 0.f);
  k_propagate<<<(unsigned)N_EDGES, 256, 0, stream>>>(hbuf, ei, deg, agg, H2, N_EDGES);
  k_selfloop_bias<<<nb((long long)N_NODES * H2, T), T, 0, stream>>>(agg, hbuf, deg, b2, N_NODES, H2);
  k_fill_f32<<<nb(H2, T), T, 0, stream>>>(colsum, H2, 0.f);
  k_fill_f32<<<nb(H2, T), T, 0, stream>>>(colsq, H2, 0.f);
  {
    dim3 g((H2 + 255) / 256, 64);
    k_colstats<<<g, 256, 0, stream>>>(agg, colsum, colsq, N_NODES, H2, (N_NODES + 63) / 64);
  }
  k_bn_relu_f32<<<nb((long long)N_NODES * H2, T), T, 0, stream>>>(agg, colsum, colsq, g2, be2,
                                                                  hbuf, N_NODES, H2);

  // ----- global max pool -----
  k_fill_f32<<<nb((long long)NBATCH * H2, T), T, 0, stream>>>(pooled, (long long)NBATCH * H2, 0.f);
  k_pool_max<<<nb((long long)N_NODES * H2, T), T, 0, stream>>>(hbuf, batch, pooled, N_NODES, H2);

  // ----- MLP head -----
  k_fc<<<nb((long long)NBATCH * FC1, T), T, 0, stream>>>(pooled, Wf1, bf1, z1, NBATCH, H2, FC1);
  k_bn_relu_small<<<FC1, 64, 0, stream>>>(z1, g5, be5, NBATCH, FC1);
  k_fc<<<nb((long long)NBATCH * FC2, T), T, 0, stream>>>(z1, Wf2, bf2, z2, NBATCH, FC1, FC2);
  k_bn_relu_small<<<FC2, 64, 0, stream>>>(z2, g6, be6, NBATCH, FC2);
  k_fc<<<nb((long long)NBATCH * EMB, T), T, 0, stream>>>(z2, Wf3, bf3, z3, NBATCH, FC2, EMB);
  k_l2norm<<<NBATCH, EMB, 0, stream>>>(z3, out);
}
